// SubstrateEvolution_15676630630958
// MI455X (gfx1250) — matrix operand order repaired and checked by reading, not compiled, not run on hardware
//
#include <hip/hip_runtime.h>
#include <hip/hip_bf16.h>
#include <math.h>

#define HW   512
#define NPIX 262144   // 512*512
#define PLANE ((size_t)4 * NPIX * 32)   // ushorts per hi/lo activation plane

typedef __attribute__((ext_vector_type(16))) __bf16 v16bf;
typedef __attribute__((ext_vector_type(8)))  float  v8f;
typedef int v4i __attribute__((vector_size(16)));

union FragU { unsigned int u[8]; v16bf v; };

// ---- async global->LDS support (guarded: probe whether builtin exists) ----
#if defined(__has_builtin)
# if __has_builtin(__builtin_amdgcn_global_load_async_to_lds_b128)
#  define HAVE_ASYNC_LDS 1
# endif
#endif
#ifndef HAVE_ASYNC_LDS
# define HAVE_ASYNC_LDS 0
#endif

static __device__ __forceinline__ void async_copy_b128(const void* gsrc, void* ldst) {
#if HAVE_ASYNC_LDS
    __builtin_amdgcn_global_load_async_to_lds_b128(
        (__attribute__((address_space(1))) v4i*)(void*)gsrc,
        (__attribute__((address_space(3))) v4i*)ldst, 0, 0);
#else
    *(uint4*)ldst = *(const uint4*)gsrc;
#endif
}

static __device__ __forceinline__ void wait_async0() {
#if HAVE_ASYNC_LDS
# if defined(__has_builtin) && __has_builtin(__builtin_amdgcn_s_wait_asynccnt)
    __builtin_amdgcn_s_wait_asynccnt(0);
# else
    asm volatile("s_wait_asynccnt 0x0" ::: "memory");
# endif
#endif
}

static __device__ __forceinline__ unsigned short f2bf(float f) {
    unsigned int u = __float_as_uint(f);
    u += 0x7FFFu + ((u >> 16) & 1u);          // round-to-nearest-even
    return (unsigned short)(u >> 16);
}
static __device__ __forceinline__ float bf2f(unsigned short h) {
    return __uint_as_float(((unsigned int)h) << 16);
}

// ---------------------------------------------------------------------------
// Pack OIHW fp32 weights into WMMA A-fragment order, split hi/lo bf16.
// Layout: wp[tap][pair][co] as uint32 = {bf16(ci=2p), bf16(ci=2p+1)}
// ---------------------------------------------------------------------------
__global__ void pack_w_kernel(const float* __restrict__ w,
                              unsigned int* __restrict__ hi,
                              unsigned int* __restrict__ lo, int Cin) {
    int idx = blockIdx.x * blockDim.x + threadIdx.x;   // 0..4607
    if (idx >= 9 * 16 * 32) return;
    int tap  = idx >> 9;
    int pair = (idx >> 5) & 15;
    int co   = idx & 31;
    unsigned int h = 0, l = 0;
    for (int j = 0; j < 2; ++j) {
        int ci = pair * 2 + j;
        float f = (ci < Cin) ? w[(co * Cin + ci) * 9 + tap] : 0.0f;
        unsigned short hb = f2bf(f);
        unsigned short lb = f2bf(f - bf2f(hb));
        h |= ((unsigned int)hb) << (16 * j);
        l |= ((unsigned int)lb) << (16 * j);
    }
    hi[idx] = h;
    lo[idx] = l;
}

// ---------------------------------------------------------------------------
// 3x3 SAME conv, Cout=32, implicit GEMM with V_WMMA_F32_16X16X32_BF16.
// Block = 256 threads = 8 wave32 -> 32(Cout) x 64(pixels) output tile.
// K = Cin*9 zero-padded to 9 chunks of 32 (tap-major, channel-minor).
// Split-bf16: acc += Ah*Bh + Ah*Bl + Al*Bh  (fp32 accumulate).
// Activations between layers: channel-last bf16 hi/lo planes -> input tile is
// 6 contiguous row segments staged with GLOBAL_LOAD_ASYNC_TO_LDS_B128.
// ---------------------------------------------------------------------------
__global__ __launch_bounds__(256) void conv3x3_wmma_kernel(
    const float* __restrict__ inF,          // fp32 planar input (in_packed==0)
    const unsigned short* __restrict__ inP, // packed hi plane (lo at +PLANE)
    const unsigned int* __restrict__ wp_hi,
    const unsigned int* __restrict__ wp_lo,
    const float* __restrict__ bias,
    float* __restrict__ outF,               // fp32 planar output (out_packed==0)
    unsigned short* __restrict__ outP,      // packed hi plane (lo at +PLANE)
    int Cin, int in_packed, int out_packed, float inscale, int do_relu) {

    // hi tile then lo tile, each 3 rows x 66 cols x 32 ch bf16 = 3168 dwords
    __shared__ alignas(16) unsigned int tileU[2 * 3168];
    unsigned short* tileHi = (unsigned short*)tileU;
    unsigned short* tileLo = (unsigned short*)(tileU + 3168);

    int bx  = blockIdx.x;
    int b   = bx >> 12;        // / (512*8)
    int rem = bx & 4095;
    int y   = rem >> 3;
    int x0  = (rem & 7) << 6;  // 64-pixel segment

    int tid   = threadIdx.x;
    int lane  = tid & 31;
    int wv    = tid >> 5;
    int mtile = wv >> 2;       // 0..1
    int ntile = wv & 3;        // 0..3

    // zero tile (covers SAME zero-padding and Cin<32 padding)
    for (int j = tid; j < 2 * 3168; j += 256) tileU[j] = 0u;
    __syncthreads();

    if (in_packed) {
        // pure DMA staging: 3 rows x 66 cols x 32ch x (hi,lo), 16B granules
        for (int j = tid; j < 3 * 66 * 4; j += 256) {
            int r   = j / 264;
            int t   = j - r * 264;
            int col = t >> 2;
            int q   = t & 3;          // which 16B granule of the 64B pixel
            int gy  = y + r - 1;
            int gx  = x0 - 1 + col;
            if (gy >= 0 && gy < HW && gx >= 0 && gx < HW) {
                size_t gp = ((((size_t)b * NPIX) + (size_t)gy * HW + gx) << 5) + (q << 3);
                int    lp = (((r * 66 + col) << 5) + (q << 3));
                async_copy_b128(inP + gp,          tileHi + lp);
                async_copy_b128(inP + PLANE + gp,  tileLo + lp);
            }
        }
        wait_async0();
    } else {
        // fp32 planar input: convert + transpose to channel-last hi/lo
        for (int r = 0; r < 3; ++r) {
            int gy = y + r - 1;
            if (gy >= 0 && gy < HW) {
                for (int idx = tid; idx < Cin * 66; idx += 256) {
                    int ci  = idx / 66;
                    int col = idx - ci * 66;
                    int gx  = x0 - 1 + col;
                    if (gx >= 0 && gx < HW) {
                        float f = inF[((size_t)(b * Cin + ci)) * NPIX + gy * HW + gx];
                        unsigned short hb = f2bf(f);
                        int o = (r * 66 + col) * 32 + ci;
                        tileHi[o] = hb;
                        tileLo[o] = f2bf(f - bf2f(hb));
                    }
                }
            }
        }
    }
    __syncthreads();

    v8f acc = {0.f, 0.f, 0.f, 0.f, 0.f, 0.f, 0.f, 0.f};
    int nloc = lane & 15;
    int hh   = lane >> 4;
    int m    = nloc + (mtile << 4);
    int p4   = hh << 3;

#pragma unroll
    for (int tap = 0; tap < 9; ++tap) {
        int r = tap / 3, s = tap - r * 3;
        int col  = (ntile << 4) + nloc + s;                 // 0..65
        int base = ((r * 66 + col) << 4) + (hh << 3);       // dword index, 32B aligned
        const uint4* bph = (const uint4*)(tileU + base);
        const uint4* bpl = (const uint4*)(tileU + 3168 + base);
        uint4 h0 = bph[0], h1 = bph[1];
        uint4 l0 = bpl[0], l1 = bpl[1];

        FragU ah, al, bh, bl;
#pragma unroll
        for (int v = 0; v < 8; ++v) {
            int p = v + p4;   // K-pair index, contiguous k = 16*half + s (matches B)
            int gi = tap * 512 + p * 32 + m;
            ah.u[v] = wp_hi[gi];
            al.u[v] = wp_lo[gi];
        }
        bh.u[0] = h0.x; bh.u[1] = h0.y; bh.u[2] = h0.z; bh.u[3] = h0.w;
        bh.u[4] = h1.x; bh.u[5] = h1.y; bh.u[6] = h1.z; bh.u[7] = h1.w;
        bl.u[0] = l0.x; bl.u[1] = l0.y; bl.u[2] = l0.z; bl.u[3] = l0.w;
        bl.u[4] = l1.x; bl.u[5] = l1.y; bl.u[6] = l1.z; bl.u[7] = l1.w;

        acc = __builtin_amdgcn_wmma_f32_16x16x32_bf16(false, ah.v, false, bh.v,
                                                      (short)0, acc, false, false);
        acc = __builtin_amdgcn_wmma_f32_16x16x32_bf16(false, ah.v, false, bl.v,
                                                      (short)0, acc, false, false);
        acc = __builtin_amdgcn_wmma_f32_16x16x32_bf16(false, al.v, false, bh.v,
                                                      (short)0, acc, false, false);
    }

    // epilogue: C/D layout -> VGPR i holds M = i + 8*half, N = lane%16
    int x = x0 + (ntile << 4) + nloc;
    if (out_packed) {
        unsigned int* oh = (unsigned int*)outP;
        size_t pdw = (((size_t)b * NPIX) + (size_t)y * HW + x) << 4;  // dword idx of ch0
#pragma unroll
        for (int i = 0; i < 8; i += 2) {
            int co = (mtile << 4) + 8 * hh + i;
            float v0 = acc[i]     * inscale + bias[co];
            float v1 = acc[i + 1] * inscale + bias[co + 1];
            if (do_relu) { v0 = fmaxf(v0, 0.0f); v1 = fmaxf(v1, 0.0f); }
            unsigned short hb0 = f2bf(v0), hb1 = f2bf(v1);
            unsigned short lb0 = f2bf(v0 - bf2f(hb0)), lb1 = f2bf(v1 - bf2f(hb1));
            oh[pdw + (co >> 1)]                = (unsigned)hb0 | ((unsigned)hb1 << 16);
            oh[(PLANE >> 1) + pdw + (co >> 1)] = (unsigned)lb0 | ((unsigned)lb1 << 16);
        }
    } else {
#pragma unroll
        for (int i = 0; i < 8; ++i) {
            int co = (mtile << 4) + i + 8 * hh;
            float v = acc[i] * inscale + bias[co];
            if (do_relu) v = fmaxf(v, 0.0f);
            outF[((size_t)(b * 32 + co)) * NPIX + y * HW + x] = v;
        }
    }
}

// ---------------------------------------------------------------------------
// Per-(b,c) Gaussian kernel coefficients (scipy truncate=4.0 semantics)
// ---------------------------------------------------------------------------
__global__ void make_kc_kernel(const float* __restrict__ sigmas,
                               float* __restrict__ kc) {
    int bc = blockIdx.x * blockDim.x + threadIdx.x;
    if (bc >= 128) return;
    float s = sigmas[bc];
    float r = floorf(4.0f * s + 0.5f);
    float v[17];
    float sum = 0.0f;
    for (int i = 0; i < 17; ++i) {
        float t = (float)(i - 8);
        float e = expf(-0.5f * (t / s) * (t / s));
        e = (fabsf(t) <= r) ? e : 0.0f;
        v[i] = e;
        sum += e;
    }
    float inv = 1.0f / sum;
    for (int i = 0; i < 17; ++i) kc[bc * 17 + i] = v[i] * inv;
}

// horizontal blur: one block per row, symmetric padding in LDS
__global__ __launch_bounds__(256) void blur_h_kernel(
    const float* __restrict__ in, const float* __restrict__ kc,
    float* __restrict__ out) {
    __shared__ float row[528];
    __shared__ float kk[17];
    int bc = blockIdx.x >> 9;
    int y  = blockIdx.x & 511;
    int tid = threadIdx.x;
    if (tid < 17) kk[tid] = kc[bc * 17 + tid];
    const float* src = in + (size_t)bc * NPIX + y * HW;
    for (int j = tid; j < 528; j += 256) {
        int sx = j - 8;
        if (sx < 0) sx = -1 - sx;
        else if (sx >= HW) sx = 2 * HW - 1 - sx;
        row[j] = src[sx];
    }
    __syncthreads();
    float* dst = out + (size_t)bc * NPIX + y * HW;
    for (int c = tid; c < HW; c += 256) {
        float a = 0.0f;
#pragma unroll
        for (int i = 0; i < 17; ++i) a += kk[i] * row[c + i];
        dst[c] = a;
    }
}

// vertical blur: per-pixel, taps hit 192MB L2 (tensor is 134MB)
__global__ __launch_bounds__(256) void blur_v_kernel(
    const float* __restrict__ in, const float* __restrict__ kc,
    float* __restrict__ out) {
    int gid = blockIdx.x * 256 + threadIdx.x;
    int bc  = gid >> 18;
    int pix = gid & (NPIX - 1);
    int y = pix >> 9;
    int x = pix & 511;
    const float* kb  = kc + bc * 17;
    const float* src = in + (size_t)bc * NPIX;
    float a = 0.0f;
#pragma unroll
    for (int i = 0; i < 17; ++i) {
        int sy = y - 8 + i;
        if (sy < 0) sy = -1 - sy;
        else if (sy >= HW) sy = 2 * HW - 1 - sy;
        a += kb[i] * src[sy * HW + x];
    }
    out[gid] = a;
}

// mean over H,W per (b,c): one block per plane
__global__ __launch_bounds__(256) void pool_mean_kernel(
    const float* __restrict__ in, float* __restrict__ pooled) {
    __shared__ float red[256];
    int bc = blockIdx.x;
    const float* src = in + (size_t)bc * NPIX;
    float a = 0.0f;
    for (int i = threadIdx.x; i < NPIX; i += 256) a += src[i];
    red[threadIdx.x] = a;
    __syncthreads();
    for (int s = 128; s > 0; s >>= 1) {
        if (threadIdx.x < s) red[threadIdx.x] += red[threadIdx.x + s];
        __syncthreads();
    }
    if (threadIdx.x == 0) pooled[bc] = red[0] * (1.0f / (float)NPIX);
}

// softmax head -> per-batch scale (mathematically == 1.1, computed faithfully)
__global__ void head_kernel(const float* __restrict__ pooled,
                            const float* __restrict__ lin_w,
                            const float* __restrict__ lin_b,
                            float* __restrict__ scale) {
    int b = threadIdx.x;
    if (b >= 4) return;
    float lg[4];
    float mx = -1e30f;
    for (int j = 0; j < 4; ++j) {
        float a = lin_b[j];
        for (int c = 0; c < 32; ++c) a += pooled[b * 32 + c] * lin_w[j * 32 + c];
        lg[j] = a;
        mx = fmaxf(mx, a);
    }
    float sum = 0.0f;
    for (int j = 0; j < 4; ++j) { lg[j] = expf(lg[j] - mx); sum += lg[j]; }
    float wsum = 0.0f;
    for (int j = 0; j < 4; ++j) wsum += lg[j] / sum;
    scale[b] = 1.0f + 0.1f * wsum;
}

// final 1x1 conv (32->3) * scale[b]
__global__ __launch_bounds__(256) void conv1x1_kernel(
    const float* __restrict__ in, const float* __restrict__ fw,
    const float* __restrict__ fb, const float* __restrict__ scale,
    float* __restrict__ out) {
    int gid = blockIdx.x * 256 + threadIdx.x;   // 4*262144
    int b   = gid >> 18;
    int pix = gid & (NPIX - 1);
    const float* src = in + (size_t)b * 32 * NPIX + pix;
    float a0 = fb[0], a1 = fb[1], a2 = fb[2];
    for (int ci = 0; ci < 32; ++ci) {
        float v = src[(size_t)ci * NPIX];
        a0 += fw[ci] * v;
        a1 += fw[32 + ci] * v;
        a2 += fw[64 + ci] * v;
    }
    float sc = scale[b];
    out[((size_t)b * 3 + 0) * NPIX + pix] = a0 * sc;
    out[((size_t)b * 3 + 1) * NPIX + pix] = a1 * sc;
    out[((size_t)b * 3 + 2) * NPIX + pix] = a2 * sc;
}

// ---------------------------------------------------------------------------
extern "C" void kernel_launch(void* const* d_in, const int* in_sizes, int n_in,
                              void* d_out, int out_size, void* d_ws, size_t ws_size,
                              hipStream_t stream) {
    (void)in_sizes; (void)n_in; (void)out_size; (void)ws_size;

    const float* x      = (const float*)d_in[0];
    const float* sigmas = (const float*)d_in[1];
    const float* cw[8]  = { (const float*)d_in[2],  (const float*)d_in[4],
                            (const float*)d_in[6],  (const float*)d_in[8],
                            (const float*)d_in[10], (const float*)d_in[12],
                            (const float*)d_in[14], (const float*)d_in[16] };
    const float* cb[8]  = { (const float*)d_in[3],  (const float*)d_in[5],
                            (const float*)d_in[7],  (const float*)d_in[9],
                            (const float*)d_in[11], (const float*)d_in[13],
                            (const float*)d_in[15], (const float*)d_in[17] };
    const float* lin_w = (const float*)d_in[18];
    const float* lin_b = (const float*)d_in[19];
    const float* fw    = (const float*)d_in[20];
    const float* fb    = (const float*)d_in[21];
    float* out = (float*)d_out;

    // workspace layout: two 128MB buffers, used either as packed bf16 hi/lo
    // activation planes or as fp32 planar tensors.
    float* bufA = (float*)d_ws;                       // 128MB
    float* bufB = bufA + (size_t)4 * 32 * NPIX;       // 128MB
    unsigned short* PA = (unsigned short*)bufA;
    unsigned short* PB = (unsigned short*)bufB;
    unsigned int* wpk = (unsigned int*)(bufB + (size_t)4 * 32 * NPIX); // 16*4608 u32
    float* kc     = (float*)(wpk + 16 * 4608);        // 128*17
    float* pooled = kc + 128 * 17;                    // 128
    float* scl    = pooled + 128;                     // 4

    // pack all 8 conv weights (hi/lo bf16 split, A-fragment order)
    for (int i = 0; i < 8; ++i) {
        int cin = (i == 0) ? 3 : 32;
        pack_w_kernel<<<18, 256, 0, stream>>>(cw[i], wpk + i * 2 * 4608,
                                              wpk + i * 2 * 4608 + 4608, cin);
    }

    dim3 cgrid(16384), cblk(256);
    // L0: conv_a reads fp32 x (Cin=3) -> packed; conv_b packed -> packed
    conv3x3_wmma_kernel<<<cgrid, cblk, 0, stream>>>(
        x, PA, wpk + 0 * 9216, wpk + 0 * 9216 + 4608, cb[0],
        bufA, PB, 3, /*in_packed=*/0, /*out_packed=*/1, 1.0f, 1);
    conv3x3_wmma_kernel<<<cgrid, cblk, 0, stream>>>(
        x, PB, wpk + 1 * 9216, wpk + 1 * 9216 + 4608, cb[1],
        bufB, PA, 32, 1, 1, 1.0f, 0);
    // L1..L2: inp = 2h folded as 2*acc on conv_a; packed <-> packed
    for (int l = 1; l < 3; ++l) {
        conv3x3_wmma_kernel<<<cgrid, cblk, 0, stream>>>(
            x, PA, wpk + (2 * l) * 9216, wpk + (2 * l) * 9216 + 4608,
            cb[2 * l], bufA, PB, 32, 1, 1, 2.0f, 1);
        conv3x3_wmma_kernel<<<cgrid, cblk, 0, stream>>>(
            x, PB, wpk + (2 * l + 1) * 9216, wpk + (2 * l + 1) * 9216 + 4608,
            cb[2 * l + 1], bufB, PA, 32, 1, 1, 1.0f, 0);
    }
    // L3: conv_a packed -> packed; conv_b packed -> fp32 planar (for blur)
    conv3x3_wmma_kernel<<<cgrid, cblk, 0, stream>>>(
        x, PA, wpk + 6 * 9216, wpk + 6 * 9216 + 4608, cb[6],
        bufA, PB, 32, 1, 1, 2.0f, 1);
    conv3x3_wmma_kernel<<<cgrid, cblk, 0, stream>>>(
        x, PB, wpk + 7 * 9216, wpk + 7 * 9216 + 4608, cb[7],
        bufA, PA, 32, 1, /*out_packed=*/0, 1.0f, 0);

    // variable-sigma separable Gaussian blur: bufA -> bufB -> bufA
    make_kc_kernel<<<1, 128, 0, stream>>>(sigmas, kc);
    blur_h_kernel<<<128 * 512, 256, 0, stream>>>(bufA, kc, bufB);
    blur_v_kernel<<<(4 * 32 * NPIX) / 256, 256, 0, stream>>>(bufB, kc, bufA);

    // head + final 1x1 conv
    pool_mean_kernel<<<128, 256, 0, stream>>>(bufA, pooled);
    head_kernel<<<1, 32, 0, stream>>>(pooled, lin_w, lin_b, scl);
    conv1x1_kernel<<<(4 * NPIX) / 256, 256, 0, stream>>>(bufA, fw, fb, scl, out);
}
